// NeuralCDE_4054449127599
// MI455X (gfx1250) — matrix operand order repaired and checked by reading, not compiled, not run on hardware
//
#include <hip/hip_runtime.h>

// ---------------------------------------------------------------------------
// Neural CDE forward for MI455X (gfx1250, wave32, WMMA).
//
// B=256 batch, P=511 RK4 steps, C=32 channels, H=256 hidden, O=10 out.
// Grid: 256 WGs = 16 batch-groups x 16 h-parts. Each group of 16 WGs owns a
// 16-row batch tile; the big h2 @ W3^T GEMM is split by h across the group,
// h1/h2 GEMMs are recomputed redundantly per WG (cheap), and the per-eval
// k = (f(z)+b3).dX partials are exchanged through L2 with an atomic barrier.
// All GEMMs use v_wmma_f32_16x16x32_f16 with f32 accumulation.
//
// Round-2 -> Round-3: kill the remaining scratch spill. (a) evc/evw local
// arrays (runtime-indexed -> private memory) replaced by ternaries; (b) all
// loop-invariant per-lane addresses and bias scalars hoisted above the
// 511-step loop so inner loads/stores are base + immediate only.
// ---------------------------------------------------------------------------

#define Bq 256
#define Cq 32
#define Hq 256
#define Oq 10
#define Pq 511          // T - 1 spline pieces
#define MT 16           // batch rows per group
#define SPLIT 16        // workgroups cooperating per group
#define NGROUP (Bq / MT)
#define ZST 264         // LDS f16 row stride (256 + 8 pad, keeps 16B align)

typedef __attribute__((ext_vector_type(16))) _Float16 v16h;
typedef __attribute__((ext_vector_type(8)))  _Float16 v8h;
typedef __attribute__((ext_vector_type(8)))  float    v8f;

// Workspace layout (bytes)
#define OFF_W1H   0u
#define OFF_W2H   131072u
#define OFF_W3H   262144u
#define OFF_WIH   4456448u
#define OFF_KBUF  4472832u   // [2][NGROUP][MT][Hq] f32
#define OFF_ZFIN  4997120u   // [Bq][Hq] f32
#define OFF_CNT   5259264u   // [NGROUP*64] u32 (cacheline padded)

__device__ __forceinline__ v8f wmma16(v16h a, v16h b, v8f c) {
  return __builtin_amdgcn_wmma_f32_16x16x32_f16(
      /*neg_a=*/false, a, /*neg_b=*/false, b,
      /*c_mod=*/(short)0, c, /*reuse_a=*/false, /*reuse_b=*/false);
}

// A fragment at a precomputed per-lane address (16x32 f16 tile).
// ISA 7.12.2: lane(l): M=l&15, hi=l>>4; elems 0..7 -> K = hi*8 + 0..7,
// elems 8..15 -> K = 16 + hi*8 + 0..7  => two 16B chunks 16 halves apart.
__device__ __forceinline__ v16h fragA_at(const _Float16* p) {
  v8h c0 = *(const v8h*)(p);
  v8h c1 = *(const v8h*)(p + 16);
  return __builtin_shufflevector(c0, c1, 0,1,2,3,4,5,6,7,8,9,10,11,12,13,14,15);
}

// B fragment at a precomputed per-lane address (B^T rows; lane: n=l&15,
// hi=l>>4, elems 0..7 -> K = hi*8 + 0..7, elems 8..15 -> K = 16 + hi*8 + 0..7).
__device__ __forceinline__ v16h fragB_at(const _Float16* p) {
  v8h c0 = *(const v8h*)(p);
  v8h c1 = *(const v8h*)(p + 16);
  return __builtin_shufflevector(c0, c1, 0,1,2,3,4,5,6,7,8,9,10,11,12,13,14,15);
}

__device__ __forceinline__ void group_barrier(unsigned* cnt, unsigned target) {
  __syncthreads();                       // all prior kbuf stores issued
  if (threadIdx.x == 0) {
    __hip_atomic_fetch_add(cnt, 1u, __ATOMIC_RELEASE, __HIP_MEMORY_SCOPE_AGENT);
    while (__hip_atomic_load(cnt, __ATOMIC_ACQUIRE, __HIP_MEMORY_SCOPE_AGENT) < target)
      __builtin_amdgcn_s_sleep(2);       // low-power spin on L2 line
  }
  __syncthreads();                       // acquire-invalidate covers whole WGP$
}

__device__ __forceinline__ float elu1(float x) {
  return x > 0.0f ? x : (__expf(x) - 1.0f);
}

// ---------------------------------------------------------------------------
__global__ void __launch_bounds__(256)
cde_prep(const float* __restrict__ W1, const float* __restrict__ W2,
         const float* __restrict__ W3, const float* __restrict__ Wi,
         _Float16* __restrict__ W1h, _Float16* __restrict__ W2h,
         _Float16* __restrict__ W3h, _Float16* __restrict__ Wih,
         unsigned* __restrict__ cnt) {
  size_t i = (size_t)blockIdx.x * blockDim.x + threadIdx.x;
  if (i < (size_t)Hq * Hq)      { W1h[i] = (_Float16)W1[i]; W2h[i] = (_Float16)W2[i]; }
  if (i < (size_t)Hq * Cq * Hq) { W3h[i] = (_Float16)W3[i]; }
  if (i < (size_t)Hq * Cq)      { Wih[i] = (_Float16)Wi[i]; }
  if (i < NGROUP * 64)          { cnt[i] = 0u; }
}

// ---------------------------------------------------------------------------
__global__ void __launch_bounds__(256)
__attribute__((amdgpu_waves_per_eu(2)))
cde_main(const float* __restrict__ coeffs,
         const _Float16* __restrict__ W1h, const _Float16* __restrict__ W2h,
         const _Float16* __restrict__ W3h, const _Float16* __restrict__ Wih,
         const float* __restrict__ b1, const float* __restrict__ b2,
         const float* __restrict__ b3, const float* __restrict__ bi,
         float* __restrict__ kbuf, float* __restrict__ zfinal,
         unsigned* __restrict__ cnt) {
  __shared__ __align__(16) _Float16 lz [MT * ZST];   // z_eval (f16)
  __shared__ __align__(16) _Float16 lh1[MT * ZST];   // elu(z@W1^T+b1) (f16)
  __shared__ __align__(16) _Float16 lh2[MT * ZST];   // elu(h1@W2^T+b2) (f16)
  __shared__ float zf[MT * Hq];                      // z state (f32)
  __shared__ float dxl[3][MT * Cq];                  // dX at f=0,0.5,1

  const int tid   = threadIdx.x;
  const int lane  = tid & 31;
  const int wave  = tid >> 5;
  const int part  = blockIdx.x & (SPLIT - 1);        // h-block owner
  const int group = blockIdx.x >> 4;                 // batch tile
  const int n  = lane & 15;
  const int hi = lane >> 4;

  unsigned* bar = cnt + group * 64;
  unsigned  bar_t = 0;

  // per-thread z ownership: 256 threads x 16 elements = 16x256
  const int zrow = tid >> 4;
  const int zcol = (tid & 15) << 4;
  float zacc[16];

  // ---- z_init = a0 @ Wi^T + bi -----------------------------------------
  #pragma unroll
  for (int i = 0; i < 2; ++i) {
    int idx = tid * 2 + i, r = idx >> 5, ch = idx & 31;
    lz[r * ZST + ch] =
        (_Float16)coeffs[(size_t)(group * MT + r) * Pq * (4 * Cq) + ch];
  }
  __syncthreads();
  {
    v16h af = fragA_at(lz + n * ZST + hi * 8);       // K = 32 exactly
    #pragma unroll
    for (int t = 0; t < 2; ++t) {
      int nt = wave * 2 + t;
      v8f acc = {};
      acc = wmma16(af, fragB_at(Wih + (nt * 16 + n) * Cq + hi * 8), acc);
      int ncol = nt * 16 + n;
      float bb = bi[ncol];
      #pragma unroll
      for (int r = 0; r < 8; ++r) zf[(r + 8 * hi) * Hq + ncol] = acc[r] + bb;
    }
  }
  __syncthreads();

  // ---- hoisted loop-invariant per-lane addresses / scalars ---------------
  const _Float16* lzA  = lz  + n * ZST + hi * 8;     // A-frag bases (LDS)
  const _Float16* lh1A = lh1 + n * ZST + hi * 8;
  const _Float16* lh2A = lh2 + n * ZST + hi * 8;
  const _Float16* w1b = W1h + (size_t)(wave * 32 + n) * Hq + hi * 8;
  const _Float16* w2b = W2h + (size_t)(wave * 32 + n) * Hq + hi * 8;
  const _Float16* w3b =
      W3h + (size_t)((part * 16 + wave * 2) * Cq + n) * Hq + hi * 8;
  const float* dxb0 = &dxl[0][(8 * hi) * Cq + n];
  const float* dxb1 = &dxl[1][(8 * hi) * Cq + n];
  const float* dxb2 = &dxl[2][(8 * hi) * Cq + n];
  _Float16* sh1 = lh1 + (8 * hi) * ZST + wave * 32 + n;   // h-tile store bases
  _Float16* sh2 = lh2 + (8 * hi) * ZST + wave * 32 + n;
  float* kwb0 = kbuf + (size_t)group * (MT * Hq)
                     + (8 * hi) * Hq + part * 16 + wave * 2;
  float* kwb1 = kwb0 + (size_t)NGROUP * (MT * Hq);
  const float* krb0 = kbuf + (size_t)group * (MT * Hq) + zrow * Hq + zcol;
  const float* krb1 = krb0 + (size_t)NGROUP * (MT * Hq);
  float*    zfp = zf + zrow * Hq + zcol;
  _Float16* lzp = lz + zrow * ZST + zcol;

  const float bb1a = b1[wave * 32 + n];
  const float bb1b = b1[wave * 32 + 16 + n];
  const float bb2a = b2[wave * 32 + n];
  const float bb2b = b2[wave * 32 + 16 + n];
  const float bb3_00 = b3[(part * 16 + wave * 2 + 0) * Cq + n];
  const float bb3_01 = b3[(part * 16 + wave * 2 + 0) * Cq + 16 + n];
  const float bb3_10 = b3[(part * 16 + wave * 2 + 1) * Cq + n];
  const float bb3_11 = b3[(part * 16 + wave * 2 + 1) * Cq + 16 + n];

  #pragma unroll
  for (int i = 0; i < 16; ++i) {
    zacc[i] = 0.0f;
    lzp[i] = (_Float16)zfp[i];
  }

  for (int p = 0; p < Pq; ++p) {
    // dX/dt at f = 0, 0.5, 1 for this spline piece
    #pragma unroll
    for (int i = 0; i < 2; ++i) {
      int idx = tid * 2 + i, r = idx >> 5, ch = idx & 31;
      const float* cp =
          coeffs + ((size_t)(group * MT + r) * Pq + p) * (4 * Cq);
      float bv = cp[Cq + ch], cv = cp[2 * Cq + ch], dv = cp[3 * Cq + ch];
      dxl[0][r * Cq + ch] = bv;
      dxl[1][r * Cq + ch] = bv + cv + 0.75f * dv;
      dxl[2][r * Cq + ch] = bv + 2.0f * cv + 3.0f * dv;
    }

    for (int e = 0; e < 4; ++e) {
      __syncthreads();                               // lz / dxl ready

      // ---- h1 = elu(z_eval @ W1^T + b1) ------------------------------
      {
        v8f acc0 = {}, acc1 = {};
        #pragma unroll
        for (int hf = 0; hf < 2; ++hf) {
          v16h af[4];
          #pragma unroll
          for (int kk = 0; kk < 4; ++kk)
            af[kk] = fragA_at(lzA + (hf * 4 + kk) * 32);
          #pragma unroll
          for (int kk = 0; kk < 4; ++kk) {
            const int kt = hf * 4 + kk;
            acc0 = wmma16(af[kk], fragB_at(w1b + kt * 32), acc0);
            acc1 = wmma16(af[kk], fragB_at(w1b + 16 * Hq + kt * 32), acc1);
          }
        }
        #pragma unroll
        for (int r = 0; r < 8; ++r) {
          sh1[r * ZST]      = (_Float16)elu1(acc0[r] + bb1a);
          sh1[r * ZST + 16] = (_Float16)elu1(acc1[r] + bb1b);
        }
      }
      __syncthreads();

      // ---- h2 = elu(h1 @ W2^T + b2) ----------------------------------
      {
        v8f acc0 = {}, acc1 = {};
        #pragma unroll
        for (int hf = 0; hf < 2; ++hf) {
          v16h af[4];
          #pragma unroll
          for (int kk = 0; kk < 4; ++kk)
            af[kk] = fragA_at(lh1A + (hf * 4 + kk) * 32);
          #pragma unroll
          for (int kk = 0; kk < 4; ++kk) {
            const int kt = hf * 4 + kk;
            acc0 = wmma16(af[kk], fragB_at(w2b + kt * 32), acc0);
            acc1 = wmma16(af[kk], fragB_at(w2b + 16 * Hq + kt * 32), acc1);
          }
        }
        #pragma unroll
        for (int r = 0; r < 8; ++r) {
          sh2[r * ZST]      = (_Float16)elu1(acc0[r] + bb2a);
          sh2[r * ZST + 16] = (_Float16)elu1(acc1[r] + bb2b);
        }
      }
      __syncthreads();

      // ---- k partial: for owned h, ((h2 @ W3^T) + b3) . dX -----------
      const float* dxe = (e == 0) ? dxb0 : ((e == 3) ? dxb2 : dxb1);
      float* kw = (e & 1) ? kwb1 : kwb0;
      #pragma unroll
      for (int hh = 0; hh < 2; ++hh) {
        v8f acc0 = {}, acc1 = {};                    // c-tiles 0..15 / 16..31
        #pragma unroll
        for (int hf = 0; hf < 2; ++hf) {
          v16h af[4];
          #pragma unroll
          for (int kk = 0; kk < 4; ++kk)
            af[kk] = fragA_at(lh2A + (hf * 4 + kk) * 32);
          #pragma unroll
          for (int kk = 0; kk < 4; ++kk) {
            const int kt = hf * 4 + kk;
            acc0 = wmma16(af[kk],
                          fragB_at(w3b + (hh * 32 + 0)  * Hq + kt * 32), acc0);
            acc1 = wmma16(af[kk],
                          fragB_at(w3b + (hh * 32 + 16) * Hq + kt * 32), acc1);
          }
        }
        const float bbA = hh ? bb3_10 : bb3_00;
        const float bbB = hh ? bb3_11 : bb3_01;
        float s[8];
        #pragma unroll
        for (int r = 0; r < 8; ++r)
          s[r] = (acc0[r] + bbA) * dxe[r * Cq] +
                 (acc1[r] + bbB) * dxe[r * Cq + 16];
        #pragma unroll
        for (int off = 1; off < 16; off <<= 1) {
          #pragma unroll
          for (int r = 0; r < 8; ++r) s[r] += __shfl_xor(s[r], off, 32);
        }
        if (n == 0) {
          #pragma unroll
          for (int r = 0; r < 8; ++r) kw[r * Hq + hh] = s[r];
        }
      }

      // ---- exchange k across the 16 cooperating WGs ------------------
      bar_t += SPLIT;
      group_barrier(bar, bar_t);

      const float* kr = (e & 1) ? krb1 : krb0;
      if (e < 3) {
        const float ec = (e == 2) ? 1.0f : 0.5f;     // z_eval coefficient
        const float ew = (e == 0) ? 1.0f : 2.0f;     // RK4 weight
        #pragma unroll
        for (int i = 0; i < 16; ++i) {
          float kv = kr[i];
          zacc[i] += ew * kv;
          lzp[i] = (_Float16)(zfp[i] + ec * kv);
        }
      } else {
        #pragma unroll
        for (int i = 0; i < 16; ++i) {
          float kv = kr[i];
          float zn = zfp[i] + (zacc[i] + kv) * (1.0f / 6.0f);
          zfp[i] = zn;
          lzp[i] = (_Float16)zn;
          zacc[i] = 0.0f;
        }
      }
    }
  }

  __syncthreads();
  if (part == 0) {
    float* zo = zfinal + (size_t)(group * MT + zrow) * Hq + zcol;
    #pragma unroll
    for (int i = 0; i < 16; ++i) zo[i] = zfp[i];
  }
}

// ---------------------------------------------------------------------------
__global__ void __launch_bounds__(256)
cde_readout(const float* __restrict__ zfinal, const float* __restrict__ Wr,
            const float* __restrict__ br, float* __restrict__ out) {
  int i = blockIdx.x * blockDim.x + threadIdx.x;
  if (i >= Bq * Oq) return;
  int b = i / Oq, o = i % Oq;
  float s = br[o];
  #pragma unroll 8
  for (int k = 0; k < Hq; ++k) s += zfinal[b * Hq + k] * Wr[o * Hq + k];
  out[i] = s;
}

// ---------------------------------------------------------------------------
extern "C" void kernel_launch(void* const* d_in, const int* in_sizes, int n_in,
                              void* d_out, int out_size, void* d_ws, size_t ws_size,
                              hipStream_t stream) {
  (void)in_sizes; (void)n_in; (void)out_size; (void)ws_size;
  const float* coeffs = (const float*)d_in[1];
  const float* Wi = (const float*)d_in[2];
  const float* bi = (const float*)d_in[3];
  const float* W1 = (const float*)d_in[4];
  const float* b1 = (const float*)d_in[5];
  const float* W2 = (const float*)d_in[6];
  const float* b2 = (const float*)d_in[7];
  const float* W3 = (const float*)d_in[8];
  const float* b3 = (const float*)d_in[9];
  const float* Wr = (const float*)d_in[10];
  const float* br = (const float*)d_in[11];

  char* ws = (char*)d_ws;
  _Float16* W1h = (_Float16*)(ws + OFF_W1H);
  _Float16* W2h = (_Float16*)(ws + OFF_W2H);
  _Float16* W3h = (_Float16*)(ws + OFF_W3H);
  _Float16* Wih = (_Float16*)(ws + OFF_WIH);
  float*    kb  = (float*)(ws + OFF_KBUF);
  float*    zfn = (float*)(ws + OFF_ZFIN);
  unsigned* cnt = (unsigned*)(ws + OFF_CNT);

  // 1) fp16 weight conversion + barrier-counter reset (stream-ordered)
  int prep_blocks = (Hq * Cq * Hq + 255) / 256;      // 8192
  cde_prep<<<prep_blocks, 256, 0, stream>>>(W1, W2, W3, Wi, W1h, W2h, W3h, Wih, cnt);

  // 2) persistent 511-step RK4 integration, 16 groups x 16 cooperating WGs
  cde_main<<<NGROUP * SPLIT, 256, 0, stream>>>(coeffs, W1h, W2h, W3h, Wih,
                                               b1, b2, b3, bi, kb, zfn, cnt);

  // 3) readout z_T @ Wr^T + br
  cde_readout<<<(Bq * Oq + 255) / 256, 256, 0, stream>>>(zfn, Wr, br, (float*)d_out);
}
